// MSA_3736621548297
// MI455X (gfx1250) — compile-verified, matrix operand order checked by reading, not run on hardware
//
#include <hip/hip_runtime.h>

typedef __attribute__((ext_vector_type(16))) __bf16 v16bf;
typedef __attribute__((ext_vector_type(8)))  float  v8f;

#define B_    8
#define N_    1024
#define DIM_  768
#define H_    12
#define DH_   64
#define INNER_ (H_ * DH_)          // 768
#define NCOLQ_ (3 * INNER_)        // 2304
#define ONE_   (B_ * H_ * N_ * DH_) // elements per q/k/v bf16 buffer
#define SCALE_ 0.125f              // DH^-0.5

union ABFrag {
    uint4 u[2];
    v16bf v;
};

__device__ __forceinline__ unsigned short f2bf(float f) {
    unsigned int u = __float_as_uint(f);
    u += 0x7FFFu + ((u >> 16) & 1u);           // round-to-nearest-even
    return (unsigned short)(u >> 16);
}

__device__ __forceinline__ v8f wmma_bf16(const ABFrag& a, const ABFrag& b, v8f c) {
    return __builtin_amdgcn_wmma_f32_16x16x32_bf16(false, a.v, false, b.v,
                                                   (short)0, c, false, false);
}

// gfx1250 async DMA: global -> LDS, 16B per lane, tracked by ASYNCcnt.
// GVS mode: vdst = per-lane LDS byte address, vaddr = 32-bit byte offset,
// saddr = uniform 64-bit base.
__device__ __forceinline__ void async_g2l_b128(const void* lds_dst,
                                               const void* base,
                                               unsigned byte_off) {
    unsigned lds = (unsigned)(size_t)lds_dst;  // flat->LDS: low 32 bits
    asm volatile("global_load_async_to_lds_b128 %0, %1, %2"
                 :: "v"(lds), "v"(byte_off), "s"(base) : "memory");
}

__device__ __forceinline__ void wait_asynccnt0() {
    asm volatile("s_wait_asynccnt 0x0" ::: "memory");
}

// ---------------------------------------------------------------------------
// Kernel 1: qkv = x @ w_qkv ; scatter bf16 into q/k/v laid out (B,H,N,DH)
// Block: 256 thr (8 waves), C tile 128x64, K step 32.
// ---------------------------------------------------------------------------
__global__ __launch_bounds__(256) void qkv_gemm(const float* __restrict__ x,
                                                const float* __restrict__ w,
                                                unsigned short* __restrict__ qkv) {
    __shared__ unsigned short sA[128 * 32];
    __shared__ unsigned short sB[64 * 32];   // transposed: [n][k]
    const int tid  = threadIdx.x;
    const int lane = tid & 31;
    const int wv   = tid >> 5;
    const int wm   = wv & 3;                 // 4 waves along M
    const int wn   = wv >> 2;                // 2 waves along N
    const int nl   = lane & 15;
    const int half = lane >> 4;
    const int row0 = blockIdx.x * 128;
    const int col0 = blockIdx.y * 64;

    v8f acc[2][2] = {};

    for (int k0 = 0; k0 < DIM_; k0 += 32) {
        __syncthreads();
        // stage A (128x32 fp32 -> bf16), 1024 float4 slots
#pragma unroll
        for (int i = 0; i < 4; ++i) {
            int s = tid + i * 256;
            int r = s >> 3, cq = s & 7;
            float4 v = *(const float4*)(x + (size_t)(row0 + r) * DIM_ + k0 + cq * 4);
            unsigned long long pk = (unsigned long long)f2bf(v.x)
                                  | ((unsigned long long)f2bf(v.y) << 16)
                                  | ((unsigned long long)f2bf(v.z) << 32)
                                  | ((unsigned long long)f2bf(v.w) << 48);
            *(unsigned long long*)(&sA[r * 32 + cq * 4]) = pk;
        }
        // stage B transposed (32x64 fp32 -> sB[n][k] bf16), 512 float4 slots
#pragma unroll
        for (int i = 0; i < 2; ++i) {
            int s = tid + i * 256;
            int kr = s >> 4, cq = s & 15;
            float4 v = *(const float4*)(w + (size_t)(k0 + kr) * NCOLQ_ + col0 + cq * 4);
            sB[(cq * 4 + 0) * 32 + kr] = f2bf(v.x);
            sB[(cq * 4 + 1) * 32 + kr] = f2bf(v.y);
            sB[(cq * 4 + 2) * 32 + kr] = f2bf(v.z);
            sB[(cq * 4 + 3) * 32 + kr] = f2bf(v.w);
        }
        __syncthreads();

        ABFrag a[2], b[2];
#pragma unroll
        for (int mt = 0; mt < 2; ++mt) {
            const uint4* p = (const uint4*)(&sA[(wm * 32 + mt * 16 + nl) * 32]);
            a[mt].u[0] = p[half];
            a[mt].u[1] = p[2 + half];
        }
#pragma unroll
        for (int nt = 0; nt < 2; ++nt) {
            const uint4* p = (const uint4*)(&sB[(wn * 32 + nt * 16 + nl) * 32]);
            b[nt].u[0] = p[half];
            b[nt].u[1] = p[2 + half];
        }
#pragma unroll
        for (int mt = 0; mt < 2; ++mt)
#pragma unroll
            for (int nt = 0; nt < 2; ++nt)
                acc[mt][nt] = wmma_bf16(a[mt], b[nt], acc[mt][nt]);
    }

    // epilogue: scatter bf16 into q/k/v (B,H,N,DH)
#pragma unroll
    for (int mt = 0; mt < 2; ++mt)
#pragma unroll
        for (int nt = 0; nt < 2; ++nt)
#pragma unroll
            for (int r = 0; r < 8; ++r) {
                int gm = row0 + wm * 32 + mt * 16 + r + 8 * half;
                int gc = col0 + wn * 32 + nt * 16 + nl;
                int which = gc / INNER_;
                int rem = gc - which * INNER_;
                int h = rem >> 6, d = rem & 63;
                int bb = gm >> 10, nn = gm & 1023;
                qkv[(size_t)which * ONE_ + ((size_t)((bb * H_ + h) << 10) + nn) * DH_ + d]
                    = f2bf(acc[mt][nt][r]);
            }
}

// ---------------------------------------------------------------------------
// Kernel 2: flash attention per (b,h, 64-query block). 128 thr (4 waves).
// ---------------------------------------------------------------------------
__global__ __launch_bounds__(128) void attn_kernel(const unsigned short* __restrict__ qkv,
                                                   const float* __restrict__ bias_table,
                                                   unsigned short* __restrict__ attn) {
    __shared__ unsigned short sK[64 * 64];   // [key][d]
    __shared__ unsigned short sV[64 * 64];   // transposed: [d][key]
    __shared__ unsigned short sP[64 * 64];   // [row][key]
    __shared__ float sBias[128];             // bias values for this (qb, j0) block
    const int tid  = threadIdx.x;
    const int lane = tid & 31;
    const int wv   = tid >> 5;               // 0..3
    const int nl   = lane & 15;
    const int half = lane >> 4;

    const int qb = blockIdx.x & 15;
    const int bh = blockIdx.x >> 4;
    const int h  = bh % H_;
    const int bb = bh / H_;

    const unsigned short* qp = qkv + (size_t)bh * N_ * DH_;
    const unsigned short* kp = qp + ONE_;
    const unsigned short* vp = qp + 2 * (size_t)ONE_;
    const float* bt = bias_table + h * (2 * N_ - 1);

    const int i0 = qb * 64 + wv * 16;

    // preload Q A-fragments (rows are contiguous 128B in global)
    ABFrag aq[2];
    {
        const uint4* p = (const uint4*)(qp + (size_t)(i0 + nl) * DH_);
#pragma unroll
        for (int kc = 0; kc < 2; ++kc) {
            aq[kc].u[0] = p[kc * 4 + half];
            aq[kc].u[1] = p[kc * 4 + 2 + half];
        }
    }

    v8f o[4] = {};
    float mrow[8], lrow[8];
#pragma unroll
    for (int r = 0; r < 8; ++r) { mrow[r] = -1e30f; lrow[r] = 0.f; }

    for (int j0 = 0; j0 < N_; j0 += 64) {
        __syncthreads();
        // stage K tile via async DMA: the 64x64 bf16 tile is 8KB contiguous
        {
            const unsigned short* kbase = kp + (size_t)j0 * DH_;
#pragma unroll
            for (int i = 0; i < 4; ++i) {
                int s = tid + i * 128;          // 512 x 16B
                async_g2l_b128(&((uint4*)sK)[s], kbase, (unsigned)(s * 16));
            }
        }
        // stage bias: off = i-j spans [qb*64 - j0 - 63, qb*64 - j0 + 63]
        {
            int off = qb * 64 - j0 - 63 + tid;       // 128 values
            int idx = (off <= 0) ? -off : (N_ - 1 + off);
            sBias[tid] = bt[idx];
        }
        // stage V tile transposed: sV[d][key]
#pragma unroll
        for (int i = 0; i < 8; ++i) {
            int s = tid + i * 128;           // 1024 uint2 slots
            int r = s >> 4, d4 = s & 15;
            uint2 w = *(const uint2*)(vp + (size_t)(j0 + r) * DH_ + d4 * 4);
            sV[(d4 * 4 + 0) * 64 + r] = (unsigned short)(w.x & 0xFFFFu);
            sV[(d4 * 4 + 1) * 64 + r] = (unsigned short)(w.x >> 16);
            sV[(d4 * 4 + 2) * 64 + r] = (unsigned short)(w.y & 0xFFFFu);
            sV[(d4 * 4 + 3) * 64 + r] = (unsigned short)(w.y >> 16);
        }
        wait_asynccnt0();
        __syncthreads();

        // S = Q K^T (16x64 per wave): 8 WMMAs
        v8f s4[4] = {};
#pragma unroll
        for (int kc = 0; kc < 2; ++kc)
#pragma unroll
            for (int ct = 0; ct < 4; ++ct) {
                ABFrag bk;
                const uint4* p = (const uint4*)(sK + (ct * 16 + nl) * 64);
                bk.u[0] = p[kc * 4 + half];
                bk.u[1] = p[kc * 4 + 2 + half];
                s4[ct] = wmma_bf16(aq[kc], bk, s4[ct]);
            }

        // scale + relative position bias from LDS:
        // sBias index = (i - j) - offmin = wv*16 + r + 8*half - ct*16 - nl + 63
#pragma unroll
        for (int ct = 0; ct < 4; ++ct)
#pragma unroll
            for (int r = 0; r < 8; ++r) {
                int bi = wv * 16 + r + 8 * half - ct * 16 - nl + 63;
                s4[ct][r] = s4[ct][r] * SCALE_ + sBias[bi];
            }

        // online softmax (rows live within 16-lane halves of the wave)
#pragma unroll
        for (int r = 0; r < 8; ++r) {
            float mx = fmaxf(fmaxf(s4[0][r], s4[1][r]), fmaxf(s4[2][r], s4[3][r]));
#pragma unroll
            for (int d = 1; d < 16; d <<= 1)
                mx = fmaxf(mx, __shfl_xor(mx, d, 32));
            float mnew = fmaxf(mrow[r], mx);
            float corr = __expf(mrow[r] - mnew);
            mrow[r] = mnew;
            float rs = 0.f;
#pragma unroll
            for (int ct = 0; ct < 4; ++ct) {
                float pv = __expf(s4[ct][r] - mnew);
                s4[ct][r] = pv;
                rs += pv;
            }
#pragma unroll
            for (int d = 1; d < 16; d <<= 1)
                rs += __shfl_xor(rs, d, 32);
            lrow[r] = lrow[r] * corr + rs;
#pragma unroll
            for (int ct = 0; ct < 4; ++ct)
                o[ct][r] *= corr;
        }

        // P (C-layout) -> LDS bf16 -> A-fragment layout
#pragma unroll
        for (int ct = 0; ct < 4; ++ct)
#pragma unroll
            for (int r = 0; r < 8; ++r)
                sP[(wv * 16 + r + 8 * half) * 64 + ct * 16 + nl] = f2bf(s4[ct][r]);
        __syncthreads();

        // O += P @ V : 8 WMMAs
#pragma unroll
        for (int kc = 0; kc < 2; ++kc) {
            ABFrag ap;
            const uint4* pp = (const uint4*)(sP + (wv * 16 + nl) * 64);
            ap.u[0] = pp[kc * 4 + half];
            ap.u[1] = pp[kc * 4 + 2 + half];
#pragma unroll
            for (int ct = 0; ct < 4; ++ct) {
                ABFrag bv;
                const uint4* pv = (const uint4*)(sV + (ct * 16 + nl) * 64);
                bv.u[0] = pv[kc * 4 + half];
                bv.u[1] = pv[kc * 4 + 2 + half];
                o[ct] = wmma_bf16(ap, bv, o[ct]);
            }
        }
    }

    // normalize + store bf16 in (B,N,H,DH) so the projection reads row-major
#pragma unroll
    for (int r = 0; r < 8; ++r) {
        float inv = 1.f / lrow[r];
        int i = i0 + r + 8 * half;
#pragma unroll
        for (int ct = 0; ct < 4; ++ct) {
            int d = ct * 16 + nl;
            attn[(((size_t)bb * N_ + i) * H_ + h) * DH_ + d] = f2bf(o[ct][r] * inv);
        }
    }
}

// ---------------------------------------------------------------------------
// Kernel 3: out = attn @ w_out + b_out (A already bf16). Same 128x64 skeleton,
// A-tile staged with async global->LDS DMA.
// ---------------------------------------------------------------------------
__global__ __launch_bounds__(256) void out_gemm(const unsigned short* __restrict__ a,
                                                const float* __restrict__ w,
                                                const float* __restrict__ bias,
                                                float* __restrict__ out) {
    __shared__ unsigned short sA[128 * 32];
    __shared__ unsigned short sB[64 * 32];
    const int tid  = threadIdx.x;
    const int lane = tid & 31;
    const int wv   = tid >> 5;
    const int wm   = wv & 3;
    const int wn   = wv >> 2;
    const int nl   = lane & 15;
    const int half = lane >> 4;
    const int row0 = blockIdx.x * 128;
    const int col0 = blockIdx.y * 64;

    v8f acc[2][2] = {};

    for (int k0 = 0; k0 < INNER_; k0 += 32) {
        __syncthreads();
        // stage A: straight bf16 copy via async DMA, 512 x 16B
#pragma unroll
        for (int i = 0; i < 2; ++i) {
            int s = tid + i * 256;
            int r = s >> 2, q = s & 3;
            unsigned goff = (unsigned)((((row0 + r) * INNER_ + k0) << 1) + q * 16);
            async_g2l_b128(&((uint4*)sA)[s], a, goff);
        }
        // stage B transposed fp32 -> bf16, 512 float4 slots
#pragma unroll
        for (int i = 0; i < 2; ++i) {
            int s = tid + i * 256;
            int kr = s >> 4, cq = s & 15;
            float4 v = *(const float4*)(w + (size_t)(k0 + kr) * DIM_ + col0 + cq * 4);
            sB[(cq * 4 + 0) * 32 + kr] = f2bf(v.x);
            sB[(cq * 4 + 1) * 32 + kr] = f2bf(v.y);
            sB[(cq * 4 + 2) * 32 + kr] = f2bf(v.z);
            sB[(cq * 4 + 3) * 32 + kr] = f2bf(v.w);
        }
        wait_asynccnt0();
        __syncthreads();

        ABFrag fa[2], fb[2];
#pragma unroll
        for (int mt = 0; mt < 2; ++mt) {
            const uint4* p = (const uint4*)(&sA[(wm * 32 + mt * 16 + nl) * 32]);
            fa[mt].u[0] = p[half];
            fa[mt].u[1] = p[2 + half];
        }
#pragma unroll
        for (int nt = 0; nt < 2; ++nt) {
            const uint4* p = (const uint4*)(&sB[(wn * 32 + nt * 16 + nl) * 32]);
            fb[nt].u[0] = p[half];
            fb[nt].u[1] = p[2 + half];
        }
#pragma unroll
        for (int mt = 0; mt < 2; ++mt)
#pragma unroll
            for (int nt = 0; nt < 2; ++nt)
                acc[mt][nt] = wmma_bf16(fa[mt], fb[nt], acc[mt][nt]);
    }

#pragma unroll
    for (int mt = 0; mt < 2; ++mt)
#pragma unroll
        for (int nt = 0; nt < 2; ++nt)
#pragma unroll
            for (int r = 0; r < 8; ++r) {
                int gm = row0 + wm * 32 + mt * 16 + r + 8 * half;
                int gc = col0 + wn * 32 + nt * 16 + nl;
                out[(size_t)gm * DIM_ + gc] = acc[mt][nt][r] + bias[gc];
            }
}

// ---------------------------------------------------------------------------
extern "C" void kernel_launch(void* const* d_in, const int* in_sizes, int n_in,
                              void* d_out, int out_size, void* d_ws, size_t ws_size,
                              hipStream_t stream) {
    (void)in_sizes; (void)n_in; (void)out_size; (void)ws_size;
    const float* x          = (const float*)d_in[0];
    const float* w_qkv      = (const float*)d_in[1];
    const float* bias_table = (const float*)d_in[2];
    const float* w_out      = (const float*)d_in[3];
    const float* b_out      = (const float*)d_in[4];
    // d_in[5] (bias_idx) is recomputed analytically in-kernel.

    unsigned short* qkv  = (unsigned short*)d_ws;              // 3*ONE_ bf16
    unsigned short* attn = qkv + (size_t)3 * ONE_;             // ONE_ bf16

    qkv_gemm<<<dim3(64, 36), 256, 0, stream>>>(x, w_qkv, qkv);
    attn_kernel<<<dim3(B_ * H_ * 16), 128, 0, stream>>>(qkv, bias_table, attn);
    out_gemm<<<dim3(64, 12), 256, 0, stream>>>(attn, w_out, b_out, (float*)d_out);
}
